// AWQW4A16Linear_53884659696166
// MI455X (gfx1250) — matrix operand order repaired and checked by reading, not compiled, not run on hardware
//
#include <hip/hip_runtime.h>
#include <stdint.h>

typedef __attribute__((ext_vector_type(16))) _Float16     v16h;
typedef __attribute__((ext_vector_type(2)))  _Float16     h2;
typedef __attribute__((ext_vector_type(8)))  float        v8f;
typedef __attribute__((ext_vector_type(4)))  unsigned int u32x4;
typedef __attribute__((ext_vector_type(8)))  int          i32x8;
typedef __attribute__((ext_vector_type(4)))  int          i32x4;

#define TOKENS 32
#define KDIM   4096
#define NDIM   11008
#define KP     (KDIM / 2)    // 2048 int32 per qweight row
#define NP     (NDIM / 4)    // 2752 qweight rows
#define NTILE  128           // output columns per block
#define KSTEP  128           // K per pipeline step (2 quant groups)
#define NSTEPS (KDIM / KSTEP)
#define CROWS  (KSTEP / 2)   // int32 per qweight row per step (64)

// ---- Tensor Data Mover: stage a 32-row x 64-int32 qweight tile into LDS ----
__device__ __forceinline__ void tdm_issue_qtile(uint32_t lds_off, const int* qw,
                                                int np0, int c0) {
  uint64_t ga = (uint64_t)(uintptr_t)qw + ((uint64_t)np0 * KP + (uint64_t)c0) * 4ull;
  u32x4 g0;
  g0[0] = 1u;                                                  // count=1 (valid user D#)
  g0[1] = lds_off;                                             // lds_addr (bytes)
  g0[2] = (uint32_t)ga;                                        // global_addr[31:0]
  g0[3] = ((uint32_t)(ga >> 32) & 0x01FFFFFFu) | (2u << 30);   // addr[56:32] | type=2
  i32x8 g1;
  g1[0] = (2 << 16);                                           // data_size = 4B; no multicast
  g1[1] = (int)((uint32_t)(KP & 0xFFFF) << 16);                // tensor_dim0[15:0]
  g1[2] = (int)(((uint32_t)(NP & 0xFFFF) << 16) | (uint32_t)(KP >> 16)); // dim1 lo | dim0 hi
  g1[3] = (int)(((uint32_t)CROWS << 16) | (uint32_t)(NP >> 16));         // tile_dim0 | dim1 hi
  g1[4] = 32;                                                  // tile_dim1 = 32 rows
  g1[5] = KP;                                                  // tensor_dim0_stride[31:0]
  g1[6] = 0;
  g1[7] = 0;
  i32x4 gz = {0, 0, 0, 0};
#if defined(__clang_major__) && (__clang_major__ >= 23)
  i32x8 gz8 = {0, 0, 0, 0, 0, 0, 0, 0};
  __builtin_amdgcn_tensor_load_to_lds(g0, g1, gz, gz, gz8, 0);
#else
  __builtin_amdgcn_tensor_load_to_lds(g0, g1, gz, gz, 0);
#endif
}

// ---- prepass: x fp32 -> f16 (x is tiny: 256 KB, reused by every block) ----
__global__ void awq_cvt_x_f16_kernel(const float* __restrict__ x,
                                     _Float16* __restrict__ xh) {
  int i = blockIdx.x * 256 + threadIdx.x;
  xh[i] = (_Float16)x[i];
}

__global__ __launch_bounds__(256, 2)
void awq_w4a16_wmma_kernel(const _Float16* __restrict__ xh,
                           const int* __restrict__ qw,
                           const float* __restrict__ wscales,
                           const float* __restrict__ wzeros,
                           const float* __restrict__ bias,
                           float* __restrict__ out) {
  __shared__ int qtile[2][32 * CROWS];  // double-buffered weight stage, 2 x 8 KB

  const int tid    = threadIdx.x;
  const int wave   = tid >> 5;
  const int lane   = tid & 31;
  const int lane16 = lane & 15;
  const int hi     = lane >> 4;                       // 0 or 1 (lane half)

  const int n0       = blockIdx.x * NTILE;
  const int n        = n0 + wave * 16 + lane16;       // this lane's output column
  const int np0      = n0 >> 2;                       // qweight row-tile base
  const int np_local = (wave * 16 + lane16) >> 2;     // 0..31 within tile
  const int jsh      = (lane16 & 3) * 8;              // nibble shift base = 8*j

  const uint32_t lds0 = (uint32_t)(uintptr_t)(&qtile[0][0]);
  const uint32_t lds1 = (uint32_t)(uintptr_t)(&qtile[1][0]);

  if (wave == 0) tdm_issue_qtile(lds0, qw, np0, 0);

  v8f acc0 = {};   // tokens 0..15
  v8f acc1 = {};   // tokens 16..31

  for (int ks = 0; ks < NSTEPS; ++ks) {
    const int cur = ks & 1;
    if (wave == 0) {
      if (ks + 1 < NSTEPS) {
        tdm_issue_qtile(cur ? lds0 : lds1, qw, np0, (ks + 1) * CROWS);
        __builtin_amdgcn_s_wait_tensorcnt(1);   // current tile landed; prefetch in flight
      } else {
        __builtin_amdgcn_s_wait_tensorcnt(0);
      }
    }
    __syncthreads();

    const int k0 = ks * KSTEP;

    // ---- batched A-fragment loads for the whole step (one clause of 16 b128;
    //      latency hidden behind the dequant VALU work below) ----
    // 16-bit A layout: lane<16 -> K {0..7,16..23}; lane>=16 -> K {8..15,24..31}
    const _Float16* xbase = xh + (size_t)lane16 * KDIM + k0 + hi * 8;
    v16h a0[4], a1[4];
#pragma unroll
    for (int ch = 0; ch < 4; ++ch) {
      const _Float16* xr = xbase + ch * 32;
      ((i32x4*)&a0[ch])[0] = *(const i32x4*)(xr);
      ((i32x4*)&a0[ch])[1] = *(const i32x4*)(xr + 16);
      ((i32x4*)&a1[ch])[0] = *(const i32x4*)(xr + 16 * KDIM);
      ((i32x4*)&a1[ch])[1] = *(const i32x4*)(xr + 16 * KDIM + 16);
    }

    // ---- per-group dequant constants (f16 0x6400 trick):
    //      w = (q - z)*s = (1024+q)*s + (-(1024+z)*s) ----
    const int gbase = k0 / 64;
    const float s0f = wscales[(size_t)gbase * NDIM + n];
    const float z0f = wzeros [(size_t)gbase * NDIM + n];
    const float s1f = wscales[(size_t)(gbase + 1) * NDIM + n];
    const float z1f = wzeros [(size_t)(gbase + 1) * NDIM + n];
    const _Float16 s0h = (_Float16)s0f, s1h = (_Float16)s1f;
    const _Float16 c0h = (_Float16)(-(1024.0f + z0f) * s0f);
    const _Float16 c1h = (_Float16)(-(1024.0f + z1f) * s1f);
    const h2 s2_0 = {s0h, s0h}, c2_0 = {c0h, c0h};
    const h2 s2_1 = {s1h, s1h}, c2_1 = {c1h, c1h};

    const int* qrowbase = &qtile[cur][np_local * CROWS];

#pragma unroll
    for (int ch = 0; ch < 4; ++ch) {           // four k=32 WMMA chunks per step
      const h2 s2 = (ch < 2) ? s2_0 : s2_1;
      const h2 c2 = (ch < 2) ? c2_0 : c2_1;

      // ---- B fragment: lane = column n, K {0..7,16..23} / {8..15,24..31} per lane half ----
      const int  cb = ch * 16 + hi * 4;        // int32 index within row
      i32x4 w0 = *(const i32x4*)(qrowbase + cb);
      i32x4 w1 = *(const i32x4*)(qrowbase + cb + 8);
      uint32_t bh32[8] __attribute__((aligned(32)));
#pragma unroll
      for (int m = 0; m < 4; ++m) {
        uint32_t u = ((uint32_t)w0[m]) >> jsh;
        uint32_t p = (u & 0xFu) | ((u << 12) & 0x000F0000u) | 0x64006400u;
        h2 r = __builtin_bit_cast(h2, p) * s2 + c2;        // v_pk_fma_f16
        bh32[m] = __builtin_bit_cast(uint32_t, r);
        u = ((uint32_t)w1[m]) >> jsh;
        p = (u & 0xFu) | ((u << 12) & 0x000F0000u) | 0x64006400u;
        r = __builtin_bit_cast(h2, p) * s2 + c2;
        bh32[4 + m] = __builtin_bit_cast(uint32_t, r);
      }
      const v16h bv = *(const v16h*)bh32;

      acc0 = __builtin_amdgcn_wmma_f32_16x16x32_f16(false, a0[ch], false, bv,
                                                    (short)0, acc0, false, false);
      acc1 = __builtin_amdgcn_wmma_f32_16x16x32_f16(false, a1[ch], false, bv,
                                                    (short)0, acc1, false, false);
    }
    __syncthreads();   // protect buffer about to be overwritten by next prefetch
  }

  // ---- epilogue: C/D layout => M = r + 8*hi (+16 for second tile), N = lane16 ----
  const float bb = bias[n];
#pragma unroll
  for (int r = 0; r < 8; ++r) {
    const int m0 = hi * 8 + r;
    out[(size_t)m0 * NDIM + n]        = acc0[r] + bb;
    out[(size_t)(m0 + 16) * NDIM + n] = acc1[r] + bb;
  }
}

extern "C" void kernel_launch(void* const* d_in, const int* in_sizes, int n_in,
                              void* d_out, int out_size, void* d_ws, size_t ws_size,
                              hipStream_t stream) {
  (void)in_sizes; (void)n_in; (void)out_size; (void)ws_size;
  const float* x      = (const float*)d_in[0];
  const int*   qw     = (const int*)d_in[1];
  const float* scales = (const float*)d_in[2];
  const float* zeros  = (const float*)d_in[3];
  const float* bias   = (const float*)d_in[4];
  float*       out    = (float*)d_out;
  _Float16*    xh     = (_Float16*)d_ws;   // 32*4096 f16 = 256 KB scratch

  awq_cvt_x_f16_kernel<<<(TOKENS * KDIM) / 256, 256, 0, stream>>>(x, xh);
  awq_w4a16_wmma_kernel<<<NDIM / NTILE, 256, 0, stream>>>(xh, qw, scales, zeros,
                                                          bias, out);
}
